// ScaledDotProductAttention_60919816127088
// MI455X (gfx1250) — compile-verified, matrix operand order checked by reading, not run on hardware
//
#include <hip/hip_runtime.h>
#include <hip/hip_bf16.h>

typedef __attribute__((ext_vector_type(16))) __bf16 v16bf;
typedef __attribute__((ext_vector_type(8)))  float  v8f;
typedef int v4i_ __attribute__((vector_size(16)));   // 128-bit payload type for async LDS builtin

#define B_    2
#define H_    16
#define S_    2048
#define D_    128
#define QT    16          // query rows per workgroup
#define SC_LD 2052        // scores LDS row stride (f32): 2052 % 64 == 4 -> conflict-free
#define QS_LD 136         // Q tile LDS row stride (bf16): 68 dwords % 64 == 4
#define VC_LD 132         // V chunk LDS row stride (f32): 8*132 % 64 == 32 -> halves conflict-free
#define SCALE 0.08838834764831845f   // 1/sqrt(128)

#define SC_BYTES  (QT * SC_LD * 4)            // 131328
#define QS_BYTES  (QT * QS_LD * 2)            // 4352
#define VBUF_OFF  (SC_BYTES + QS_BYTES)       // 135680 (16B aligned)
#define VBUF_ELEM (32 * VC_LD)                // floats per V buffer
#define SMEM_BYTES (VBUF_OFF + 2 * VBUF_ELEM * 4)   // 169472 < 320KB WGP LDS

// ---- CDNA5 async global->LDS path (ASYNCcnt), with guarded fallback ----
#if __has_builtin(__builtin_amdgcn_global_load_async_to_lds_b128)
#define HAS_ASYNC_LDS 1
#else
#define HAS_ASYNC_LDS 0
#endif

__device__ __forceinline__ void stage16(const float* g, float* l) {
#if HAS_ASYNC_LDS
    __builtin_amdgcn_global_load_async_to_lds_b128(
        (__attribute__((address_space(1))) v4i_*)(unsigned long long)g,
        (__attribute__((address_space(3))) v4i_*)(unsigned int)(unsigned long long)l,
        0, 0);
#else
    *(float4*)l = *(const float4*)g;          // sync fallback: b128 load + ds_store_b128
#endif
}

#if HAS_ASYNC_LDS
#if __has_builtin(__builtin_amdgcn_s_wait_asynccnt)
#define WAIT_ASYNC(n) __builtin_amdgcn_s_wait_asynccnt(n)
#else
#define WAIT_ASYNC(n) asm volatile("s_wait_asynccnt %0" ::"i"(n) : "memory")
#endif
#else
#define WAIT_ASYNC(n) ((void)0)
#endif

__global__ __launch_bounds__(256) void attn16_kernel(
    const float* __restrict__ q,
    const float* __restrict__ k,
    const float* __restrict__ v,
    const unsigned char* __restrict__ mask,
    float* __restrict__ ctx_out,
    float* __restrict__ w_out)
{
    extern __shared__ char smem_raw[];
    float*  sc   = (float*)smem_raw;                      // [QT][SC_LD] f32 scores/weights
    __bf16* qs   = (__bf16*)(smem_raw + SC_BYTES);        // [QT][QS_LD] bf16 Q tile
    float*  vbuf = (float*)(smem_raw + VBUF_OFF);         // 2 x [32][VC_LD] f32 V chunks

    const int tid  = (int)threadIdx.x;
    const int lane = tid & 31;
    const int wv   = tid >> 5;        // wave 0..7
    const int half = lane >> 4;       // 0: lanes 0-15, 1: lanes 16-31
    const int l16  = lane & 15;

    const int q0 = (int)blockIdx.x * QT;
    const int bh = (int)blockIdx.y;
    const size_t rowbase = (size_t)bh * S_;

    // ---------------- Phase 0: Q tile -> LDS (bf16) ----------------
    {
        const float* qg = q + (rowbase + (size_t)q0) * D_;
        #pragma unroll
        for (int i = 0; i < (QT * D_) / 256; ++i) {
            int idx = tid + 256 * i;
            int r = idx >> 7;
            int c = idx & 127;
            qs[r * QS_LD + c] = (__bf16)qg[(size_t)r * D_ + c];
        }
    }
    __syncthreads();

    // ---------------- Phase 1: S = (Q K^T) * scale  (WMMA bf16) ----------------
    for (int jt = 0; jt < 16; ++jt) {
        const int j = wv * 16 + jt;
        const float* kg = k + (rowbase + (size_t)j * 16) * D_;
        if (jt + 1 < 16) {  // prefetch next K tile (8KB tile, 32 lanes x 256B)
            const float* nk = k + (rowbase + (size_t)(j + 1) * 16) * D_;
            __builtin_prefetch(nk + lane * 64, 0, 1);
        }
        v8f acc = {};
        #pragma unroll
        for (int ks = 0; ks < 4; ++ks) {
            const int k0 = ks * 32;
            v16bf a, b;
            const __bf16* ap = qs + l16 * QS_LD + k0 + half * 8;
            const float*  bp = kg + (size_t)l16 * D_ + k0 + half * 8;
            #pragma unroll
            for (int e = 0; e < 8; ++e) {
                a[e]     = ap[e];
                a[e + 8] = ap[e + 16];
                b[e]     = (__bf16)bp[e];
                b[e + 8] = (__bf16)bp[e + 16];
            }
            acc = __builtin_amdgcn_wmma_f32_16x16x32_bf16(false, a, false, b,
                                                          (short)0, acc, false, false);
        }
        #pragma unroll
        for (int r = 0; r < 8; ++r)
            sc[(r + 8 * half) * SC_LD + j * 16 + l16] = acc[r] * SCALE;
    }
    __syncthreads();

    // Kick off async stage of V chunk 0 into buffer 0; it completes under phase 2.
    {
        const float* vg = v + rowbase * D_;
        #pragma unroll
        for (int i = 0; i < 4; ++i) {
            int idx = tid + 256 * i;     // float4 index within 32x128 chunk
            int r = idx >> 5;
            int c4 = idx & 31;
            stage16(vg + (size_t)r * D_ + c4 * 4, vbuf + r * VC_LD + c4 * 4);
        }
    }

    // ---------------- Phase 2: masked softmax + write weights (dominant HBM traffic) ----------------
    for (int rr = wv * 2; rr < wv * 2 + 2; ++rr) {
        const unsigned char* mrow = mask + (rowbase + (size_t)(q0 + rr)) * S_;
        float* srow = sc + rr * SC_LD;

        float m = -__builtin_inff();
        #pragma unroll 4
        for (int i = 0; i < S_ / 32; ++i) {
            int c = lane + 32 * i;
            float s = mrow[c] ? srow[c] : -__builtin_inff();
            m = fmaxf(m, s);
        }
        #pragma unroll
        for (int off = 16; off > 0; off >>= 1)
            m = fmaxf(m, __shfl_xor(m, off, 32));

        float sum = 0.f;
        #pragma unroll 4
        for (int i = 0; i < S_ / 32; ++i) {
            int c = lane + 32 * i;
            float e = mrow[c] ? __expf(srow[c] - m) : 0.f;
            srow[c] = e;
            sum += e;
        }
        #pragma unroll
        for (int off = 16; off > 0; off >>= 1)
            sum += __shfl_xor(sum, off, 32);
        const float inv = 1.0f / sum;

        float* wrow = w_out + (rowbase + (size_t)(q0 + rr)) * S_;
        #pragma unroll 4
        for (int i = 0; i < S_ / 32; ++i) {
            int c = lane + 32 * i;
            float wgt = srow[c] * inv;
            srow[c] = wgt;               // normalized weights stay in LDS for phase 3
            wrow[c] = wgt;               // coalesced 128B/wave store
        }
    }

    // ---------------- Phase 3: context = W x V, double-buffered async V staging ----------------
    {
        const int n = wv;                // this wave's 16-wide output column tile
        v8f acc = {};

        auto consume = [&](int kk) {
            const float* vs = vbuf + (kk & 1) * VBUF_ELEM + n * 16 + l16;
            const float* ap = sc + l16 * SC_LD + kk * 32 + half * 8;
            v16bf a, b;
            #pragma unroll
            for (int e = 0; e < 8; ++e) {
                a[e]     = (__bf16)ap[e];
                a[e + 8] = (__bf16)ap[e + 16];
                b[e]     = (__bf16)vs[(half * 8 + e) * VC_LD];
                b[e + 8] = (__bf16)vs[(16 + half * 8 + e) * VC_LD];
            }
            acc = __builtin_amdgcn_wmma_f32_16x16x32_bf16(false, a, false, b,
                                                          (short)0, acc, false, false);
        };

        for (int kk = 0; kk < S_ / 32 - 1; ++kk) {
            __syncthreads();             // buffer (kk+1)&1 fully consumed (also phase2->3 sync at kk=0)
            {
                const float* vg = v + (rowbase + (size_t)(kk + 1) * 32) * D_;
                float* dst = vbuf + ((kk + 1) & 1) * VBUF_ELEM;
                #pragma unroll
                for (int i = 0; i < 4; ++i) {
                    int idx = tid + 256 * i;
                    int r = idx >> 5;
                    int c4 = idx & 31;
                    stage16(vg + (size_t)r * D_ + c4 * 4, dst + r * VC_LD + c4 * 4);
                }
            }
            WAIT_ASYNC(4);               // all but the 4 just-issued ops done -> chunk kk complete
            __syncthreads();             // chunk kk visible to all waves
            consume(kk);
        }
        WAIT_ASYNC(0);
        __syncthreads();
        consume(S_ / 32 - 1);

        #pragma unroll
        for (int r = 0; r < 8; ++r) {
            int mr = r + 8 * half;
            ctx_out[(rowbase + (size_t)(q0 + mr)) * D_ + n * 16 + l16] = acc[r];
        }
    }
}

extern "C" void kernel_launch(void* const* d_in, const int* in_sizes, int n_in,
                              void* d_out, int out_size, void* d_ws, size_t ws_size,
                              hipStream_t stream) {
    (void)in_sizes; (void)n_in; (void)out_size; (void)d_ws; (void)ws_size;
    const float* q = (const float*)d_in[0];
    const float* k = (const float*)d_in[1];
    const float* v = (const float*)d_in[2];
    const unsigned char* mask = (const unsigned char*)d_in[3];

    float* ctx = (float*)d_out;                               // [B,H,S,D]
    float* w   = (float*)d_out + (size_t)B_ * H_ * S_ * D_;   // [B,H,S,S]

    static bool attr_set = false;
    if (!attr_set) {
        (void)hipFuncSetAttribute((const void*)attn16_kernel,
                                  hipFuncAttributeMaxDynamicSharedMemorySize, (int)SMEM_BYTES);
        attr_set = true;
    }

    dim3 grid(S_ / QT, B_ * H_);
    attn16_kernel<<<grid, 256, (size_t)SMEM_BYTES, stream>>>(q, k, v, mask, ctx, w);
}
